// FNO2d_25915832664152
// MI455X (gfx1250) — matrix operand order repaired and checked by reading, compiled, not run on hardware
//
#include <hip/hip_runtime.h>
#include <hip/hip_bf16.h>
#include <cmath>

// ---------------------------------------------------------------------------
// FNO2d forward for MI455X (gfx1250, wave32, WMMA).
// Spectral convs are truncated-DFT GEMMs (MODES=12 << 128). All heavy GEMMs:
// v_wmma_f32_16x16x32_f16 with multi-accumulator waves; B operands staged to
// LDS with global_load_async_to_lds_b128 (ASYNCcnt) and read back via ds ops.
// ---------------------------------------------------------------------------

typedef __attribute__((ext_vector_type(16))) _Float16 v16h;
typedef __attribute__((ext_vector_type(8)))  float    v8f;

#define WAVE 32

__device__ __forceinline__ float gelu_f(float v) {
  return 0.5f * v * (1.0f + erff(v * 0.70710678118654752440f));
}

// A fragment: 16x32 f16, row-major source A[row][k], lda elements.
// ISA layout: lane m = lane&15, half = lane>>4; VGPR r holds K pair
// k0 = 16*(r>>2) + 2*(r&3) + 8*half.
__device__ __forceinline__ v16h load_fragA_rm(const _Float16* __restrict__ A,
                                              int lda, int row0, int kOff, int lane) {
  const _Float16* p = A + (size_t)(row0 + (lane & 15)) * lda + kOff;
  int half8 = (lane >> 4) << 3;
  v16h f;
#pragma unroll
  for (int r = 0; r < 8; ++r) {
    int k0 = 16 * (r >> 2) + 2 * (r & 3) + half8;
    f[2 * r]     = p[k0];
    f[2 * r + 1] = p[k0 + 1];
  }
  return f;
}

// A fragment where the K axis is channel-strided (activations stored [C][HW]).
__device__ __forceinline__ v16h load_fragA_chan(const _Float16* __restrict__ Y,
                                                size_t chanStride, int cBase,
                                                size_t pBase, int kOff, int lane) {
  size_t p = pBase + (size_t)(lane & 15);
  int half8 = (lane >> 4) << 3;
  v16h f;
#pragma unroll
  for (int r = 0; r < 8; ++r) {
    int k0 = 16 * (r >> 2) + 2 * (r & 3) + half8 + kOff;
    f[2 * r]     = Y[(size_t)(cBase + k0) * chanStride + p];
    f[2 * r + 1] = Y[(size_t)(cBase + k0 + 1) * chanStride + p];
  }
  return f;
}

// B fragment: 32x16 f16 from Bt[n][k] (transpose of math B, ldb elements).
// Lane n = lane&15 holds K 8*(lane>>4)..+7 then 16+8*(lane>>4)..+7.
__device__ __forceinline__ v16h load_fragB(const _Float16* Bt, int ldb,
                                           int col0, int kOff, int lane) {
  const _Float16* p = Bt + (size_t)(col0 + (lane & 15)) * ldb + kOff + ((lane >> 4) << 3);
  v16h f;
#pragma unroll
  for (int e = 0; e < 16; ++e) f[e] = p[e < 8 ? e : e + 8];
  return f;
}

__device__ __forceinline__ v8f wmma_step(v16h a, v16h b, v8f c) {
  return __builtin_amdgcn_wmma_f32_16x16x32_f16(false, a, false, b, (short)0, c,
                                                false, false);
}

// ---------------------------------------------------------------------------
// Wide GEMM with LDS-staged B (async copy path):
//   block = 8 waves, computes 128 rows x (NSUB*16) cols.
//   B tile (packed Bt, ldb == K) staged via global_load_async_to_lds_b128.
// EPI: 0 = f32 store, 1 = bias+GELU -> f16, 2 = bias -> f16.
// ACHAN: A is channel-strided activations [C][16384].
// Requires: M % 128 == 0, K % 32 == 0, gridDim.x == (M/128)*nGroups.
// ---------------------------------------------------------------------------
template <int K, int NSUB, int EPI, int ACHAN>
__global__ void gemm_ldsB_kernel(const _Float16* __restrict__ A, int lda,
                                 const _Float16* __restrict__ Bt,
                                 const float* __restrict__ bias,
                                 float* __restrict__ Cf32,
                                 _Float16* __restrict__ Of16,
                                 int ldo, int M, int nGroups) {
  __shared__ _Float16 Bs[NSUB * 16 * K];   // sole LDS object -> offset 0
  const int tid = threadIdx.x;
  const int lane = tid & (WAVE - 1);
  const int wave = tid / WAVE;
  const int mblk = blockIdx.x / nGroups;
  const int ng   = blockIdx.x % nGroups;

  // ---- async stage of the B tile (contiguous: ldb == K) ----
  constexpr int BYTES = NSUB * 16 * K * 2;           // multiple of 4096
  const char* gsrc = (const char*)(Bt + (size_t)ng * (NSUB * 16) * K);
#pragma unroll
  for (int oB = tid * 16; oB < BYTES; oB += 256 * 16) {
    unsigned ldsAddr = (unsigned)oB;                 // Bs at LDS offset 0
    asm volatile("global_load_async_to_lds_b128 %0, %1, off"
                 :: "v"(ldsAddr), "v"(gsrc + oB) : "memory");
  }
  asm volatile("s_wait_asynccnt 0" ::: "memory");
  __syncthreads();

  // ---- compute: one A fragment feeds NSUB WMMAs ----
  const int m0 = (mblk * 8 + wave) * 16;
  v8f acc[NSUB];
#pragma unroll
  for (int s = 0; s < NSUB; ++s) acc[s] = (v8f){0.f,0.f,0.f,0.f,0.f,0.f,0.f,0.f};

  for (int k = 0; k < K; k += 32) {
    v16h a;
    if (ACHAN)
      a = load_fragA_chan(A, 16384, (m0 >> 14) * 32, (size_t)(m0 & 16383), k, lane);
    else
      a = load_fragA_rm(A, lda, m0, k, lane);
#pragma unroll
    for (int s = 0; s < NSUB; ++s) {
      v16h b = load_fragB(Bs, K, s * 16, k, lane);   // ds_load path
      acc[s] = wmma_step(a, b, acc[s]);
    }
  }

  // ---- epilogue ----
  const int mb = m0 + ((lane >> 4) << 3);
#pragma unroll
  for (int s = 0; s < NSUB; ++s) {
    int n = ng * (NSUB * 16) + s * 16 + (lane & 15);
    if (EPI == 0) {
#pragma unroll
      for (int r = 0; r < 8; ++r)
        Cf32[(size_t)(mb + r) * ldo + n] = acc[s][r];
    } else {
      float bn = bias[n];
#pragma unroll
      for (int r = 0; r < 8; ++r) {
        float v = acc[s][r] + bn;
        if (EPI == 1) v = gelu_f(v);
        Of16[(size_t)(mb + r) * ldo + n] = (_Float16)v;
      }
    }
  }
}

// ---------------------------------------------------------------------------
// fc0: y[b,c,h,w] = sum_j x[b,h,w,j]*W[j,c] + b[c]   (K=4, VALU)
// ---------------------------------------------------------------------------
__global__ void fc0_kernel(const float* __restrict__ x, const float* __restrict__ W,
                           const float* __restrict__ bia,
                           float* __restrict__ y, _Float16* __restrict__ y16) {
  size_t i = (size_t)blockIdx.x * blockDim.x + threadIdx.x;  // B*32*16384
  if (i >= (size_t)8 * 32 * 16384) return;
  int p = (int)(i & 16383);
  int c = (int)((i >> 14) & 31);
  int b = (int)(i >> 19);
  const float* xp = x + ((((size_t)b << 14) + p) << 2);
  float v = bia[c] + xp[0] * W[c] + xp[1] * W[32 + c] + xp[2] * W[64 + c] +
            xp[3] * W[96 + c];
  y[i] = v;
  y16[i] = (_Float16)v;
}

// ---------------------------------------------------------------------------
// Twiddle tables: FwT [32][128] (rows 24..31 zero): forward row-rDFT,
// n=2k -> cos, n=2k+1 -> -sin. iFwT [128][32] (cols 24..31 zero): inverse
// row transform with hermitian weights c_k = (k==0 ? 1 : 2)/W.
// ---------------------------------------------------------------------------
__global__ void twiddle_kernel(_Float16* __restrict__ FwT,
                               _Float16* __restrict__ iFwT) {
  int i = blockIdx.x * blockDim.x + threadIdx.x;
  if (i < 32 * 128) {
    int n = i >> 7, w = i & 127, k = n >> 1;
    float v = 0.f;
    if (n < 24) {
      float th = 6.283185307179586f * (float)((w * k) & 127) * (1.0f / 128.0f);
      v = (n & 1) ? -sinf(th) : cosf(th);
    }
    FwT[i] = (_Float16)v;
  }
  if (i < 128 * 32) {
    int w = i >> 5, n = i & 31, k = n >> 1;
    float v = 0.f;
    if (n < 24) {
      float c = (k == 0 ? 1.0f : 2.0f) * (1.0f / 128.0f);
      float th = 6.283185307179586f * (float)((w * k) & 127) * (1.0f / 128.0f);
      v = (n & 1) ? -c * sinf(th) : c * cosf(th);
    }
    iFwT[i] = (_Float16)v;
  }
}

__global__ void cvt_f16_kernel(const float* __restrict__ in,
                               _Float16* __restrict__ out, int n) {
  int i = blockIdx.x * blockDim.x + threadIdx.x;
  if (i < n) out[i] = (_Float16)in[i];
}

// in [K][N] f32 -> out [N][K] f16  (weight transpose for B fragments)
__global__ void transpose_cvt_kernel(const float* __restrict__ in,
                                     _Float16* __restrict__ out, int K, int N) {
  int i = blockIdx.x * blockDim.x + threadIdx.x;
  if (i < K * N) {
    int n = i / K, k = i % K;
    out[i] = (_Float16)in[(size_t)k * N + n];
  }
}

// ---------------------------------------------------------------------------
// Column DFT over H for kept rows r in {0..11}->ky=r, {12..23}->ky=104+r.
// rowft: [32768 rows (b,c,h)][32] (n=2kx re, 2kx+1 im).
// ---------------------------------------------------------------------------
__global__ void coldft_kernel(const float* __restrict__ rowft,
                              float* __restrict__ modes) {
  int i = blockIdx.x * blockDim.x + threadIdx.x;  // 8*32*24*12
  if (i >= 73728) return;
  int kx = i % 12;
  int r = (i / 12) % 24;
  int bc = i / 288;
  int ky = r < 12 ? r : 104 + r;
  const float* src = rowft + (size_t)bc * 128 * 32 + 2 * kx;
  float re = 0.f, im = 0.f;
  for (int h = 0; h < 128; ++h) {
    float th = 6.283185307179586f * (float)((h * ky) & 127) * (1.0f / 128.0f);
    float cs = cosf(th), sn = sinf(th);
    float a = src[h * 32], b = src[h * 32 + 1];
    re += a * cs + b * sn;   // exp(-i th)
    im += b * cs - a * sn;
  }
  modes[(size_t)i * 2]     = re;
  modes[(size_t)i * 2 + 1] = im;
}

// Spectral complex channel-mix: out[b,o,r,kx] = sum_i modes[b,i,r,kx]*w[...]
__global__ void specmul_kernel(const float* __restrict__ modes,
                               const float* __restrict__ specw, int layer,
                               float* __restrict__ out) {
  int i = blockIdx.x * blockDim.x + threadIdx.x;  // (b,o,r,kx)
  if (i >= 73728) return;
  int kx = i % 12;
  int r = (i / 12) % 24;
  int o = (i / 288) % 32;
  int b = i / (288 * 32);
  int part = r < 12 ? 0 : 1;
  int xx = r < 12 ? r : r - 12;
  size_t wb0 = ((((size_t)(layer * 2 + part) * 32) * 32 + o) * 12 + xx) * 24 +
               (size_t)kx * 2;
  const float* min = modes + (size_t)b * 32 * 576 + (size_t)(r * 12 + kx) * 2;
  float re = 0.f, im = 0.f;
  for (int ci = 0; ci < 32; ++ci) {
    float a = min[(size_t)ci * 576], bb2 = min[(size_t)ci * 576 + 1];
    float c = specw[wb0 + (size_t)ci * 9216];
    float d = specw[wb0 + (size_t)ci * 9216 + 1];
    re += a * c - bb2 * d;
    im += a * d + bb2 * c;
  }
  size_t oi = (size_t)(b * 32 + o) * 576 + (size_t)(r * 12 + kx) * 2;
  out[oi] = re;
  out[oi + 1] = im;
}

// Inverse column transform -> f16 rows [32768][32], cols 24..31 zero.
__global__ void invcol_kernel(const float* __restrict__ spec,
                              _Float16* __restrict__ invcol16) {
  int i = blockIdx.x * blockDim.x + threadIdx.x;  // 8*32*128 rows
  if (i >= 32768) return;
  int h = i & 127;
  int bc = i >> 7;
  const float* src = spec + (size_t)bc * 576;
  _Float16* dst = invcol16 + (size_t)i * 32;
  float re[12], im[12];
#pragma unroll
  for (int kx = 0; kx < 12; ++kx) { re[kx] = 0.f; im[kx] = 0.f; }
  for (int r = 0; r < 24; ++r) {
    int ky = r < 12 ? r : 104 + r;
    float th = 6.283185307179586f * (float)((h * ky) & 127) * (1.0f / 128.0f);
    float cs = cosf(th), sn = sinf(th);
#pragma unroll
    for (int kx = 0; kx < 12; ++kx) {
      float a = src[(r * 12 + kx) * 2], b = src[(r * 12 + kx) * 2 + 1];
      re[kx] += a * cs - b * sn;   // exp(+i th)
      im[kx] += a * sn + b * cs;
    }
  }
#pragma unroll
  for (int kx = 0; kx < 12; ++kx) {
    dst[2 * kx]     = (_Float16)(re[kx] * (1.0f / 128.0f));
    dst[2 * kx + 1] = (_Float16)(im[kx] * (1.0f / 128.0f));
  }
#pragma unroll
  for (int n = 24; n < 32; ++n) dst[n] = (_Float16)0.0f;
}

// ---------------------------------------------------------------------------
// Fused per-layer combine: conv1x1 (WMMA, 16x32 per wave, A reused) +
// spectral spatial result + grid conv + mesh conv + GELU -> ynext (f32+f16).
// Tiles: b(8) x tm(1024); grid = 1024 blocks of 8 waves.
// ---------------------------------------------------------------------------
__global__ void combine_kernel(const _Float16* __restrict__ y16,
                               const _Float16* __restrict__ wW16, int layer,
                               const float* __restrict__ spat,
                               const float* __restrict__ g_wb,
                               const float* __restrict__ g_bW,
                               const float* __restrict__ g_bb,
                               const float* __restrict__ g_cW,
                               const float* __restrict__ g_cb,
                               const float* __restrict__ x,
                               float* __restrict__ ynext,
                               _Float16* __restrict__ ynext16) {
  int lane = threadIdx.x & (WAVE - 1);
  int wave = threadIdx.x / WAVE;
  long tile = (long)blockIdx.x * 8 + wave;
  if (tile >= 8L * 1024) return;          // wave-uniform
  int b = (int)(tile / 1024);
  int tm = (int)(tile % 1024);
  const _Float16* Wl = wW16 + (size_t)layer * 1024;  // [o][i]
  v16h a = load_fragA_chan(y16, 16384, b * 32, (size_t)tm * 16, 0, lane);
  v8f acc[2];
#pragma unroll
  for (int s = 0; s < 2; ++s) {
    v16h bf = load_fragB(Wl, 32, s * 16, 0, lane);
    v8f z = (v8f){0.f,0.f,0.f,0.f,0.f,0.f,0.f,0.f};
    acc[s] = wmma_step(a, bf, z);
  }
  int pbase = tm * 16 + ((lane >> 4) << 3);
#pragma unroll
  for (int s = 0; s < 2; ++s) {
    int o = s * 16 + (lane & 15);
    int lo = layer * 32 + o;
    float bw0 = g_bW[lo * 2], bw1 = g_bW[lo * 2 + 1];
    float cw0 = g_cW[lo * 3], cw1 = g_cW[lo * 3 + 1], cw2 = g_cW[lo * 3 + 2];
    float cst = g_wb[lo] + g_bb[lo] + g_cb[lo];
#pragma unroll
    for (int r = 0; r < 8; ++r) {
      int p = pbase + r;
      int h = p >> 7, w = p & 127;
      const float* xp = x + ((((size_t)b << 14) + p) << 2);
      float v = acc[s][r] + spat[((size_t)(b * 32 + o) << 14) + p] +
                bw0 * (h * (1.0f / 127.0f)) + bw1 * (w * (1.0f / 127.0f)) +
                cw0 * xp[1] + cw1 * xp[2] + cw2 * xp[3] + cst;
      v = gelu_f(v);
      size_t oi = ((size_t)(b * 32 + o) << 14) + p;
      ynext[oi] = v;
      ynext16[oi] = (_Float16)v;
    }
  }
}

// fc5: out[p] = H3[p,:] . fc5_W + b   (N=1, VALU)
__global__ void fc5_kernel(const _Float16* __restrict__ H3,
                           const float* __restrict__ W,
                           const float* __restrict__ bia,
                           float* __restrict__ out) {
  size_t i = (size_t)blockIdx.x * blockDim.x + threadIdx.x;
  if (i >= 131072) return;
  const _Float16* p = H3 + i * 128;
  float acc = bia[0];
  for (int k = 0; k < 128; ++k) acc += (float)p[k] * W[k];
  out[i] = acc;
}

// ---------------------------------------------------------------------------
extern "C" void kernel_launch(void* const* d_in, const int* in_sizes, int n_in,
                              void* d_out, int out_size, void* d_ws, size_t ws_size,
                              hipStream_t stream) {
  const float* x      = (const float*)d_in[0];
  const float* fc0_W  = (const float*)d_in[1];
  const float* fc0_b  = (const float*)d_in[2];
  const float* spec_w = (const float*)d_in[3];
  const float* wW     = (const float*)d_in[4];
  const float* wb     = (const float*)d_in[5];
  const float* bW     = (const float*)d_in[6];
  const float* bb     = (const float*)d_in[7];
  const float* cW     = (const float*)d_in[8];
  const float* cb     = (const float*)d_in[9];
  const float* fc1_W  = (const float*)d_in[10];
  const float* fc1_b  = (const float*)d_in[11];
  const float* fc3_W  = (const float*)d_in[12];
  const float* fc3_b  = (const float*)d_in[13];
  const float* fc4_W  = (const float*)d_in[14];
  const float* fc4_b  = (const float*)d_in[15];
  const float* fc5_W  = (const float*)d_in[16];
  const float* fc5_b  = (const float*)d_in[17];
  float* out = (float*)d_out;

  char* ws = (char*)d_ws;
  size_t off = 0;
  auto carve = [&](size_t bytes) -> void* {
    void* p = ws + off;
    off = (off + bytes + 255) & ~(size_t)255;
    return p;
  };
  const size_t NACT = (size_t)8 * 32 * 16384;  // 4,194,304
  float*    yA       = (float*)carve(NACT * 4);
  float*    yB       = (float*)carve(NACT * 4);
  _Float16* y16A     = (_Float16*)carve(NACT * 2);
  _Float16* y16B     = (_Float16*)carve(NACT * 2);
  float*    spat     = (float*)carve(NACT * 4);
  float*    rowft    = (float*)carve((size_t)32768 * 32 * 4);
  float*    modes    = (float*)carve((size_t)73728 * 2 * 4);
  float*    specout  = (float*)carve((size_t)73728 * 2 * 4);
  _Float16* invcol16 = (_Float16*)carve((size_t)32768 * 32 * 2);
  _Float16* FwT      = (_Float16*)carve((size_t)32 * 128 * 2);
  _Float16* iFwT     = (_Float16*)carve((size_t)128 * 32 * 2);
  _Float16* wW16     = (_Float16*)carve((size_t)6 * 1024 * 2);
  _Float16* fc1T     = (_Float16*)carve((size_t)256 * 32 * 2);
  _Float16* fc3T     = (_Float16*)carve((size_t)128 * 256 * 2);
  _Float16* fc4T     = (_Float16*)carve((size_t)128 * 128 * 2);
  _Float16* H1       = (_Float16*)carve((size_t)131072 * 256 * 2);
  _Float16* H2       = (_Float16*)carve((size_t)131072 * 128 * 2);
  _Float16* H3       = (_Float16*)carve((size_t)131072 * 128 * 2);
  (void)ws_size; (void)in_sizes; (void)n_in; (void)out_size;

  // ---- constant tables / weight conversion ----
  twiddle_kernel<<<16, 256, 0, stream>>>(FwT, iFwT);
  cvt_f16_kernel<<<24, 256, 0, stream>>>(wW, wW16, 6 * 1024);
  transpose_cvt_kernel<<<32, 256, 0, stream>>>(fc1_W, fc1T, 32, 256);
  transpose_cvt_kernel<<<128, 256, 0, stream>>>(fc3_W, fc3T, 256, 128);
  transpose_cvt_kernel<<<64, 256, 0, stream>>>(fc4_W, fc4T, 128, 128);

  // ---- fc0 lift ----
  fc0_kernel<<<(unsigned)(NACT / 256), 256, 0, stream>>>(x, fc0_W, fc0_b, yA, y16A);

  // ---- 6 FNO layers (ping-pong) ----
  float*    ycur = yA;   _Float16* y16cur = y16A;
  float*    ynxt = yB;   _Float16* y16nxt = y16B;
  for (int l = 0; l < 6; ++l) {
    // forward row rDFT: [32768 x 128] * [128 -> 24(+pad 32)]
    gemm_ldsB_kernel<128, 2, 0, 0><<<256, 256, 0, stream>>>(
        y16cur, 128, FwT, nullptr, rowft, nullptr, 32, 32768, 1);
    coldft_kernel<<<288, 256, 0, stream>>>(rowft, modes);
    specmul_kernel<<<288, 256, 0, stream>>>(modes, spec_w, l, specout);
    invcol_kernel<<<128, 256, 0, stream>>>(specout, invcol16);
    // inverse row transform: [32768 x 24(pad 32)] * [24 -> 128]
    gemm_ldsB_kernel<32, 8, 0, 0><<<256, 256, 0, stream>>>(
        invcol16, 32, iFwT, nullptr, spat, nullptr, 128, 32768, 1);
    combine_kernel<<<1024, 256, 0, stream>>>(y16cur, wW16, l, spat, wb, bW, bb,
                                             cW, cb, x, ynxt, y16nxt);
    float* tf = ycur; ycur = ynxt; ynxt = tf;
    _Float16* th = y16cur; y16cur = y16nxt; y16nxt = th;
  }

  // ---- MLP head ----
  // fc1: [131072 x 32] -> 256, GELU (A channel-strided)
  gemm_ldsB_kernel<32, 4, 1, 1><<<4096, 256, 0, stream>>>(
      y16cur, 0, fc1T, fc1_b, nullptr, H1, 256, 131072, 4);
  // fc3: [131072 x 256] -> 128, GELU
  gemm_ldsB_kernel<256, 4, 1, 0><<<2048, 256, 0, stream>>>(
      H1, 256, fc3T, fc3_b, nullptr, H2, 128, 131072, 2);
  // fc4: [131072 x 128] -> 128, no activation
  gemm_ldsB_kernel<128, 4, 2, 0><<<2048, 256, 0, stream>>>(
      H2, 128, fc4T, fc4_b, nullptr, H3, 128, 131072, 2);
  fc5_kernel<<<512, 256, 0, stream>>>(H3, fc5_W, fc5_b, out);
}
